// SoftAttention_86414741996119
// MI455X (gfx1250) — matrix operand order repaired and checked by reading, not compiled, not run on hardware
//
#include <hip/hip_runtime.h>
#include <hip/hip_bf16.h>

typedef __attribute__((ext_vector_type(16))) _Float16 v16h;
typedef __attribute__((ext_vector_type(8)))  float    v8f;

#define B_    2
#define TQ_   1024
#define TV_   1024
#define D_    64
#define NEG_  1e9f
#define WAVES 4                 // waves per block == KV split factor
#define QPW   16                // query rows per block (one WMMA M-tile)
#define KT    32                // KV tile (one WMMA K for f16)
#define SEG   (TV_ / WAVES)     // 256 KV rows per wave
#define LDSP  68                // padded row stride (floats): 16B-aligned, bank-spread
#define LOG2E 1.4426950408889634f

__device__ __forceinline__ float dtanh(float x) {
#if __has_builtin(__builtin_amdgcn_tanhf)
  return __builtin_amdgcn_tanhf(x);                 // v_tanh_f32 (gfx1250 trans op)
#elif __has_builtin(__builtin_amdgcn_tanh_f32)
  return __builtin_amdgcn_tanh_f32(x);
#else
  float e = __builtin_amdgcn_exp2f(x * 2.8853900817779268f);   // exp(2x)
  return 1.0f - 2.0f * __builtin_amdgcn_rcpf(e + 1.0f);
#endif
}

__device__ __forceinline__ float dexp(float x) {
  return __builtin_amdgcn_exp2f(x * LOG2E);         // v_exp_f32
}

__global__ __launch_bounds__(WAVES * 32)
void SoftAttention_gfx1250_kernel(const float* __restrict__ Q,
                                  const float* __restrict__ V,
                                  const float* __restrict__ Sc,
                                  const unsigned char* __restrict__ QM,
                                  const unsigned char* __restrict__ VM,
                                  float* __restrict__ Out) {
  __shared__ float Qs[QPW][LDSP];             // 16 q rows (shared by all waves)
  __shared__ float Vs[WAVES][KT][LDSP];       // per-wave V tile
  __shared__ float Vadd[WAVES][KT];           // per-wave v_mask additive terms
  __shared__ float Osh[WAVES][QPW][D_];       // per-wave partial O
  __shared__ float Msh[WAVES][QPW];           // per-wave row max
  __shared__ float Lsh[WAVES][QPW];           // per-wave row sum

  const int tid   = threadIdx.x;
  const int lane  = tid & 31;
  const int wv    = tid >> 5;                 // wave id == KV segment id
  const int half  = lane >> 4;                // 0: lanes 0-15, 1: lanes 16-31
  const int lid16 = lane & 15;                // A row (softmax row) / B,C column

  const int b    = blockIdx.x / (TQ_ / QPW);
  const int q0   = (blockIdx.x % (TQ_ / QPW)) * QPW;  // block's 16-query tile
  const int qrow = q0 + lid16;                        // this lane's softmax row

  const float scale = Sc[0];

  // ---- stage the 16-row Q tile into LDS (256 float4 / 128 threads) ----
  {
    const float* qg = Q + ((size_t)b * TQ_ + q0) * D_;
    #pragma unroll
    for (int i = 0; i < (QPW * D_ / 4) / (WAVES * 32); ++i) {  // 2 iters
      int idx = tid + i * (WAVES * 32);
      int row = idx >> 4;                   // 16 float4 per row
      int c4  = idx & 15;
      float4 v = ((const float4*)qg)[row * (D_ / 4) + c4];
      *(float4*)&Qs[row][c4 * 4] = v;
    }
  }

  float mrun = -3.0e38f;                    // running row max (this segment)
  float lrun = 0.0f;                        // running row sum (this segment)
  v8f  cacc[4] = {};                        // 4 d-chunks of 16x16 f32 C

  // ---- flash loop over this wave's KV segment: 8 tiles of 32 ----
  for (int t = 0; t < SEG / KT; ++t) {
    const int k0 = wv * SEG + t * KT;
    __syncthreads();                        // protect Vs from previous tile's readers
    {
      const float* vg = V + ((size_t)b * TV_ + k0) * D_;
      #pragma unroll
      for (int i = 0; i < (KT * D_ / 4) / 32; ++i) {  // 16 float4 per lane
        int idx = lane + i * 32;
        int row = idx >> 4;
        int c4  = idx & 15;
        float4 v = ((const float4*)vg)[row * (D_ / 4) + c4];
        *(float4*)&Vs[wv][row][c4 * 4] = v;
      }
      Vadd[wv][lane] = VM[(size_t)b * TV_ + k0 + lane] ? 0.0f : -NEG_;
      if (t + 1 < SEG / KT)
        __builtin_prefetch(vg + KT * D_ + lane * 64, 0, 0);  // global_prefetch_b8
    }
    __syncthreads();                        // LDS stores visible before reads

    // ---- scores: S[m][k] = sum_d tanh(q+v), computed in A-matrix layout ----
    float acc[16];
    #pragma unroll
    for (int e = 0; e < 16; ++e) acc[e] = 0.0f;
    for (int d = 0; d < D_; d += 4) {
      float4 qv = *(const float4*)&Qs[lid16][d];
      #pragma unroll
      for (int e = 0; e < 16; ++e) {
        const int ke = (e & 7) + ((e >> 3) << 4) + (half << 3); // A-layout K
        float4 vv = *(const float4*)&Vs[wv][ke][d];
        acc[e] += dtanh(qv.x + vv.x) + dtanh(qv.y + vv.y)
                + dtanh(qv.z + vv.z) + dtanh(qv.w + vv.w);
      }
    }

    // ---- additive mask (causal + v_mask, exactly -1e9 like reference) ----
    float s[16];
    float tmax = -3.0e38f;
    #pragma unroll
    for (int e = 0; e < 16; ++e) {
      const int ke = (e & 7) + ((e >> 3) << 4) + (half << 3);
      float madd = (k0 + ke <= qrow) ? Vadd[wv][ke] : -NEG_;
      s[e] = scale * acc[e] + madd;
      tmax = fmaxf(tmax, s[e]);
    }
    // row spans this lane and its xor-16 partner (two K-halves)
    tmax = fmaxf(tmax, __shfl_xor(tmax, 16, 32));
    float mnew = fmaxf(mrun, tmax);
    float corr = dexp(mrun - mnew);
    float tsum = 0.0f;
    v16h a;
    #pragma unroll
    for (int e = 0; e < 16; ++e) {
      float p = dexp(s[e] - mnew);
      tsum += p;
      a[e] = (_Float16)p;                   // P in f16 A-matrix layout
    }
    tsum += __shfl_xor(tsum, 16, 32);
    lrun = lrun * corr + tsum;
    mrun = mnew;

    // ---- rescale C rows: C-layout row M = r + 8*half lives in lane M ----
    #pragma unroll
    for (int r = 0; r < 8; ++r) {
      float cr = __shfl(corr, r + (half << 3), 32);
      #pragma unroll
      for (int nd = 0; nd < 4; ++nd) cacc[nd][r] *= cr;
    }

    // ---- O_w += P(16x32) @ Vtile(32x16) per 16-wide d-chunk ----
    #pragma unroll
    for (int nd = 0; nd < 4; ++nd) {
      v16h bmat;
      #pragma unroll
      for (int e = 0; e < 16; ++e)          // B layout: K = (e&7) + 16*(e>>3) + 8*half, N = lid16
        bmat[e] = (_Float16)Vs[wv][(e & 7) + ((e >> 3) << 4) + (half << 3)][nd * 16 + lid16];
      cacc[nd] = __builtin_amdgcn_wmma_f32_16x16x32_f16(
          false, a, false, bmat, (short)0, cacc[nd], false, false);
    }
  }

  // ---- deposit this wave's partial (m, l, O) into LDS ----
  if (half == 0) {                           // row state lives in lanes 0..15
    Msh[wv][lid16] = mrun;
    Lsh[wv][lid16] = lrun;
  }
  #pragma unroll
  for (int r = 0; r < 8; ++r)
    #pragma unroll
    for (int nd = 0; nd < 4; ++nd)
      Osh[wv][r + (half << 3)][nd * 16 + lid16] = cacc[nd][r];
  __syncthreads();

  // ---- merge the 4 KV-segment partials (flash combine), mask, store ----
  // 16 rows x 64 cols = 1024 outputs / 128 threads = 8 each
  #pragma unroll
  for (int i = 0; i < (QPW * D_) / (WAVES * 32); ++i) {
    int idx = tid + i * (WAVES * 32);
    int row = idx >> 6;                      // D_ = 64 cols per row
    int col = idx & 63;
    float mstar = Msh[0][row];
    #pragma unroll
    for (int w = 1; w < WAVES; ++w) mstar = fmaxf(mstar, Msh[w][row]);
    float lsum = 0.0f, oval = 0.0f;
    #pragma unroll
    for (int w = 0; w < WAVES; ++w) {
      float cw = dexp(Msh[w][row] - mstar);
      lsum += cw * Lsh[w][row];
      oval += cw * Osh[w][row][col];
    }
    int   qg = q0 + row;
    float wq = QM[(size_t)b * TQ_ + qg] ? (1.0f / lsum) : 0.0f;
    Out[((size_t)b * TQ_ + qg) * D_ + col] = oval * wq;
  }
}

extern "C" void kernel_launch(void* const* d_in, const int* in_sizes, int n_in,
                              void* d_out, int out_size, void* d_ws, size_t ws_size,
                              hipStream_t stream) {
  const float*         Q  = (const float*)d_in[0];
  const float*         V  = (const float*)d_in[1];
  const float*         Sc = (const float*)d_in[2];
  const unsigned char* QM = (const unsigned char*)d_in[3];  // bool mask, 1B/elem
  const unsigned char* VM = (const unsigned char*)d_in[4];
  float* Out = (float*)d_out;

  dim3 grid(B_ * (TQ_ / QPW));             // 2 * 64 = 128 blocks
  dim3 block(WAVES * 32);                  // 128 threads = 4 wave32
  hipLaunchKernelGGL(SoftAttention_gfx1250_kernel, grid, block, 0, stream,
                     Q, V, Sc, QM, VM, Out);
}
